// Three_LSTMHMM_62886911148291
// MI455X (gfx1250) — matrix operand order repaired and checked by reading, not compiled, not run on hardware
//
#include <hip/hip_runtime.h>
#include <cstdint>
#include <cstddef>

// ---------------------------------------------------------------------------
// Three stacked LSTM-HMMs on MI455X (gfx1250, wave32, WMMA).
//   I=64, H=512, S=64, B=128, T=128, M=3
// Strategy:
//   * bf16 WMMA (v_wmma_f32_16x16x32_bf16) for the recurrent GEMM and the
//     emission GEMM, f32 accumulate.
//   * Recurrent weights staged ONCE per WG into 320KB-class LDS via the
//     CDNA5 async global->LDS path; per-step B-operands come from
//     ds_load_b128 (LDS latency on the T-serial critical path).
//   * Serial-T recurrence: 16 WGs per model, each owns 32 hidden units
//     (all 4 gates), synced per step with a global atomic arrive counter.
//   * HMM forward: f32 VALU exp/log, LDS-resident transition matrix.
// ---------------------------------------------------------------------------

typedef __attribute__((ext_vector_type(16))) __bf16 v16bf;
typedef __attribute__((ext_vector_type(8)))  float  v8f;

#define I_SZ 64
#define H_SZ 512
#define S_SZ 64
#define B_SZ 128
#define T_SZ 128
#define M_SZ 3
#define NSPLIT 16   // workgroups per model for the LSTM phase

// ---- workspace layout (bytes, 1KB aligned) --------------------------------
static constexpr size_t WS_ARRIVE = 0;          // 3*128 ints   (per-(m,t) arrive counters)
static constexpr size_t WS_BSUM   = 4096;       // 3*2048 f32   (b_ih+b_hh)
static constexpr size_t WS_BLIN   = 28672;      // 3*64 f32
static constexpr size_t WS_LOGPRI = 29696;      // 3*64 f32
static constexpr size_t WS_LOGT   = 30720;      // 3*64*64 f32
static constexpr size_t WS_XSWZ   = 79872;      // T*2*8 frags * 1KB = 2MB
static constexpr size_t WS_WLSWZ  = 2177024;    // 3*16*4 frags * 1KB
static constexpr size_t WS_WSWZ   = 2373632;    // 3*16*18*8 frags * 1KB = 6.75MB
static constexpr size_t WS_HSWZ   = 9451520;    // 3*128*16*8 frags * 1KB = 48MB
static constexpr size_t WS_EM     = 59783168;   // 3*T*B*S f32 = 12MB
// total ~72.4 MB

// dynamic LDS (uint4 units)
extern __shared__ uint4 dynbuf[];
#define LSTM_W_U4   (144 * 64)                 // 18kt * 8ct frags = 144KB
#define LSTM_LDS_BYTES (LSTM_W_U4 * 16 + 16 * 16 * 16 * 2)   // +8KB staging
#define EMIS_W_U4   (64 * 64)                  // 16kt * 4st frags = 64KB
#define EMIS_LDS_BYTES (EMIS_W_U4 * 16)

// ---- helpers --------------------------------------------------------------
__device__ __forceinline__ unsigned short f2bf(float f) {
    unsigned int u = __float_as_uint(f);
    unsigned int r = u + 0x7FFFu + ((u >> 16) & 1u);   // round-to-nearest-even-ish
    return (unsigned short)(r >> 16);
}

union FragU { uint4 u[2]; v16bf v; };

// fragment storage: 32 lanes * 16 bf16, lane-contiguous = 64 uint4 per frag
__device__ __forceinline__ v16bf load_frag(const uint4* base, int frag, int lane) {
    FragU f;
    f.u[0] = base[(size_t)frag * 64 + lane * 2 + 0];
    f.u[1] = base[(size_t)frag * 64 + lane * 2 + 1];
    return f.v;
}

__device__ __forceinline__ void store_frag(uint4* base, int frag, int lane,
                                           const unsigned short* vals) {
    uint4 a, b;
    a.x = (unsigned)vals[0]  | ((unsigned)vals[1]  << 16);
    a.y = (unsigned)vals[2]  | ((unsigned)vals[3]  << 16);
    a.z = (unsigned)vals[4]  | ((unsigned)vals[5]  << 16);
    a.w = (unsigned)vals[6]  | ((unsigned)vals[7]  << 16);
    b.x = (unsigned)vals[8]  | ((unsigned)vals[9]  << 16);
    b.y = (unsigned)vals[10] | ((unsigned)vals[11] << 16);
    b.z = (unsigned)vals[12] | ((unsigned)vals[13] << 16);
    b.w = (unsigned)vals[14] | ((unsigned)vals[15] << 16);
    base[(size_t)frag * 64 + lane * 2 + 0] = a;
    base[(size_t)frag * 64 + lane * 2 + 1] = b;
}

__device__ __forceinline__ v8f wmma_bf16(v16bf a, v16bf b, v8f c) {
    return __builtin_amdgcn_wmma_f32_16x16x32_bf16(
        /*neg_a=*/false, a, /*neg_b=*/false, b,
        /*c_mod=*/(short)0, c, /*reuse_a=*/false, /*reuse_b=*/false);
}

// async copy of `count` uint4 from global `src` to LDS region starting at
// dynbuf[dstBaseU4], strided across the whole block. CDNA5 TDM-class path:
// global_load_async_to_lds_b128 tracked by ASYNCcnt.
__device__ __forceinline__ void async_copy_u4(const uint4* __restrict__ src,
                                              int dstBaseU4, int count) {
    for (int idx = (int)threadIdx.x; idx < count; idx += (int)blockDim.x) {
        unsigned ldsoff = (unsigned)(unsigned long long)(const void*)&dynbuf[dstBaseU4 + idx];
        unsigned long long ga = (unsigned long long)(const void*)(src + idx);
        asm volatile("global_load_async_to_lds_b128 %0, %1, off"
                     :: "v"(ldsoff), "v"(ga) : "memory");
    }
    asm volatile("s_wait_asynccnt 0x0" ::: "memory");
}

// A-fragment K index (16-bit A 16x32 layout, ISA 7.12.2)
__device__ __forceinline__ int kIdxA(int e, int hf) {
    return (e < 8) ? (e + 8 * hf) : (e + 8 + 8 * hf);
}

__device__ __forceinline__ float sigmoidf(float x) {
    return 1.0f / (1.0f + __expf(-x));
}
__device__ __forceinline__ float fast_tanh(float x) {
    x = fminf(fmaxf(x, -15.0f), 15.0f);
    float e = __expf(-2.0f * x);
    return (1.0f - e) / (1.0f + e);
}

// ---------------------------------------------------------------------------
// Kernel 1: prep — zero counters, bias sums, log-softmax of A/prior,
// swizzle x / [W_ih;W_hh] / W_lin into WMMA fragment order (bf16).
// ---------------------------------------------------------------------------
#define PREP_N0 384      // arrive counters
#define PREP_N1 6144     // bsum
#define PREP_N2 192      // blin
#define PREP_N3 192      // log prior
#define PREP_N4 192      // logT columns
#define PREP_N5 65536    // Xswz  (2048 frags * 32 lanes)
#define PREP_N6 221184   // Wswz  (6912 frags * 32 lanes)
#define PREP_N7 6144     // WLswz (192 frags * 32 lanes)
#define PREP_TOTAL (PREP_N0+PREP_N1+PREP_N2+PREP_N3+PREP_N4+PREP_N5+PREP_N6+PREP_N7)

__global__ void prep_kernel(const float* __restrict__ x,
                            const float* __restrict__ W_ih,
                            const float* __restrict__ W_hh,
                            const float* __restrict__ b_ih,
                            const float* __restrict__ b_hh,
                            const float* __restrict__ W_lin,
                            const float* __restrict__ b_lin,
                            const float* __restrict__ A,
                            const float* __restrict__ prior,
                            char* __restrict__ ws) {
    int id = blockIdx.x * blockDim.x + threadIdx.x;

    int*   arrive = (int*)  (ws + WS_ARRIVE);
    float* bsum   = (float*)(ws + WS_BSUM);
    float* blin   = (float*)(ws + WS_BLIN);
    float* logpri = (float*)(ws + WS_LOGPRI);
    float* logT   = (float*)(ws + WS_LOGT);
    uint4* xswz   = (uint4*)(ws + WS_XSWZ);
    uint4* wswz   = (uint4*)(ws + WS_WSWZ);
    uint4* wlswz  = (uint4*)(ws + WS_WLSWZ);

    if (id < PREP_N0) { arrive[id] = 0; return; }
    id -= PREP_N0;

    if (id < PREP_N1) { bsum[id] = b_ih[id] + b_hh[id]; return; }
    id -= PREP_N1;

    if (id < PREP_N2) { blin[id] = b_lin[id]; return; }
    id -= PREP_N2;

    if (id < PREP_N3) {   // log_softmax(prior) per (m,s)
        int m = id / 64, s = id % 64;
        const float* p = prior + m * 64;
        float mx = -3.0e38f;
        for (int j = 0; j < 64; ++j) mx = fmaxf(mx, p[j]);
        float sm = 0.f;
        for (int j = 0; j < 64; ++j) sm += __expf(p[j] - mx);
        logpri[id] = p[s] - (mx + __logf(sm));
        return;
    }
    id -= PREP_N3;

    if (id < PREP_N4) {   // log_softmax(A, axis=0): normalize columns
        int m = id / 64, j = id % 64;
        const float* Am = A + m * 4096;
        float mx = -3.0e38f;
        for (int i = 0; i < 64; ++i) mx = fmaxf(mx, Am[i * 64 + j]);
        float sm = 0.f;
        for (int i = 0; i < 64; ++i) sm += __expf(Am[i * 64 + j] - mx);
        float lse = mx + __logf(sm);
        float* lt = logT + m * 4096;
        for (int i = 0; i < 64; ++i) lt[i * 64 + j] = Am[i * 64 + j] - lse;
        return;
    }
    id -= PREP_N4;

    if (id < PREP_N5) {   // Xswz: A-frags of x, frag = (t*2+kt)*8+mt
        int lane = id & 31, frag = id >> 5;
        int mt = frag & 7, rest = frag >> 3;
        int kt = rest & 1, t = rest >> 1;
        int hf = lane >> 4, r = lane & 15;
        int b = mt * 16 + r;
        unsigned short vals[16];
        #pragma unroll
        for (int e = 0; e < 16; ++e) {
            int k = kt * 32 + kIdxA(e, hf);
            vals[e] = f2bf(x[((size_t)b * T_SZ + t) * I_SZ + k]);
        }
        store_frag(xswz, frag, lane, vals);
        return;
    }
    id -= PREP_N5;

    if (id < PREP_N6) {   // Wswz: B-frags of [W_ih;W_hh], frag=((m*16+n)*18+kt)*8+ct
        int lane = id & 31, frag = id >> 5;
        int ct = frag & 7, r1 = frag >> 3;
        int kt = r1 % 18, r2 = r1 / 18;
        int wg = r2 & 15, m = r2 >> 4;
        int hf = lane >> 4, ncol = lane & 15;
        int c = ct * 16 + ncol;                         // WG-local gate column 0..127
        int grow = (c >> 5) * 512 + wg * 32 + (c & 31); // global gate row 0..2047
        unsigned short vals[16];
        #pragma unroll
        for (int e = 0; e < 16; ++e) {
            int k = kt * 32 + kIdxA(e, hf);             // 0..575
            float v = (k < I_SZ)
                        ? W_ih[((size_t)m * 2048 + grow) * I_SZ + k]
                        : W_hh[((size_t)m * 2048 + grow) * H_SZ + (k - I_SZ)];
            vals[e] = f2bf(v);
        }
        store_frag(wswz, frag, lane, vals);
        return;
    }
    id -= PREP_N6;

    if (id < PREP_N7) {   // WLswz: B-frags of W_lin, frag=(m*16+kt)*4+st
        int lane = id & 31, frag = id >> 5;
        int st = frag & 3, kt = (frag >> 2) & 15, m = frag >> 6;
        int hf = lane >> 4, s = st * 16 + (lane & 15);
        unsigned short vals[16];
        #pragma unroll
        for (int e = 0; e < 16; ++e) {
            int k = kt * 32 + kIdxA(e, hf);
            vals[e] = f2bf(W_lin[((size_t)m * S_SZ + s) * H_SZ + k]);
        }
        store_frag(wlswz, frag, lane, vals);
    }
}

// ---------------------------------------------------------------------------
// Kernel 2: persistent LSTM. grid = M*NSPLIT WGs, 512 threads (16 waves).
// Weights for the WG live in LDS (filled once via async global->LDS).
// Wave w: mt=w/2 (16 batch rows), jt=w%2 (16 units),
// tiles ct={jt,2+jt,4+jt,6+jt} -> i/f/g/o aligned per lane element.
// ---------------------------------------------------------------------------
__global__ __launch_bounds__(512) void lstm_kernel(char* __restrict__ ws) {
    const uint4* xswz = (const uint4*)(ws + WS_XSWZ);
    const uint4* wswz = (const uint4*)(ws + WS_WSWZ);
    uint4*       hswz = (uint4*)(ws + WS_HSWZ);
    const float* bsum = (const float*)(ws + WS_BSUM);
    int*         arrive = (int*)(ws + WS_ARRIVE);

    const int m = blockIdx.x >> 4;
    const int n = blockIdx.x & 15;          // hidden slice: units n*32..n*32+31
    const int wave = threadIdx.x >> 5;
    const int lane = threadIdx.x & 31;
    const int mt = wave >> 1;               // batch tile 0..7
    const int jt = wave & 1;                // unit half 0..1
    const int hf = lane >> 4;
    const int l16 = lane & 15;

    // one-time async stage of this WG's 144KB weight-fragment slice into LDS
    {
        const uint4* wsrc = wswz + (size_t)(((m * 16 + n) * 18) * 8) * 64;
        async_copy_u4(wsrc, 0, LSTM_W_U4);
    }
    unsigned short* stg = (unsigned short*)(dynbuf + LSTM_W_U4); // [16][16][16]
    __syncthreads();

    float bias[4];
    #pragma unroll
    for (int g = 0; g < 4; ++g)
        bias[g] = bsum[m * 2048 + g * 512 + n * 32 + jt * 16 + l16];

    v8f cst;
    #pragma unroll
    for (int v = 0; v < 8; ++v) cst[v] = 0.0f;

    for (int t = 0; t < T_SZ; ++t) {
        v8f a0, a1, a2, a3;
        #pragma unroll
        for (int v = 0; v < 8; ++v) { a0[v] = bias[0]; a1[v] = bias[1];
                                      a2[v] = bias[2]; a3[v] = bias[3]; }

        if (t + 1 < T_SZ)   // prefetch next step's x fragments toward WGP
            __builtin_prefetch(xswz + (size_t)(((t + 1) * 2) * 8 + mt) * 64, 0, 3);

        for (int kt = 0; kt < 18; ++kt) {
            if (t == 0 && kt >= 2) break;        // h0 == 0: skip recurrent K
            v16bf af;
            if (kt < 2)
                af = load_frag(xswz, (t * 2 + kt) * 8 + mt, lane);
            else
                af = load_frag(hswz, ((m * T_SZ + (t - 1)) * 16 + (kt - 2)) * 8 + mt, lane);

            // B operands from LDS: frag (kt*8 + ct), ct = {jt, 2+jt, 4+jt, 6+jt}
            const int bi = (kt * 8 + jt) * 64 + lane * 2;
            FragU b0, b1, b2, b3;
            b0.u[0] = dynbuf[bi +   0]; b0.u[1] = dynbuf[bi +   1];
            b1.u[0] = dynbuf[bi + 128]; b1.u[1] = dynbuf[bi + 129];
            b2.u[0] = dynbuf[bi + 256]; b2.u[1] = dynbuf[bi + 257];
            b3.u[0] = dynbuf[bi + 384]; b3.u[1] = dynbuf[bi + 385];
            a0 = wmma_bf16(af, b0.v, a0);        // gate i
            a1 = wmma_bf16(af, b1.v, a1);        // gate f
            a2 = wmma_bf16(af, b2.v, a2);        // gate g
            a3 = wmma_bf16(af, b3.v, a3);        // gate o
        }

        // LSTM cell update, fully per-lane (i/f/g/o share (b,j) per element)
        v8f hn;
        #pragma unroll
        for (int v = 0; v < 8; ++v) {
            float si = sigmoidf(a0[v]);
            float sf = sigmoidf(a1[v]);
            float tg = fast_tanh(a2[v]);
            float so = sigmoidf(a3[v]);
            float cn = sf * cst[v] + si * tg;
            cst[v] = cn;
            hn[v] = so * fast_tanh(cn);
        }

        // stage h chunk (16 b x 16 j) in LDS, C-layout -> row-major bf16
        #pragma unroll
        for (int v = 0; v < 8; ++v)
            stg[wave * 256 + (v + 8 * hf) * 16 + l16] = f2bf(hn[v]);
        __syncthreads();

        // gather A-fragment-ordered bytes: lane covers (r=l16, cols 8*hf..+7),
        // which is exactly uint4 slot jt of Hswz frag [m][t][n][mt].
        {
            const uint4* src = (const uint4*)(stg + wave * 256 + l16 * 16 + 8 * hf);
            uint4 val = *src;
            int frag = ((m * T_SZ + t) * 16 + n) * 8 + mt;
            hswz[(size_t)frag * 64 + lane * 2 + jt] = val;
        }

        // inter-WG step barrier for this model
        __threadfence();
        __syncthreads();
        if (threadIdx.x == 0) {
            int* ctr = &arrive[m * T_SZ + t];
            __hip_atomic_fetch_add(ctr, 1, __ATOMIC_RELEASE, __HIP_MEMORY_SCOPE_AGENT);
            while (__hip_atomic_load(ctr, __ATOMIC_ACQUIRE, __HIP_MEMORY_SCOPE_AGENT) < NSPLIT)
                __builtin_amdgcn_s_sleep(2);
        }
        __syncthreads();
    }
}

// ---------------------------------------------------------------------------
// Kernel 3: emission logits via WMMA + in-register log_softmax over S=64.
// grid = 3*64 WGs; W_lin fragments staged once in LDS (async), each wave
// handles one (t, mt) row tile and all 4 S tiles.
// ---------------------------------------------------------------------------
__global__ __launch_bounds__(512) void emis_kernel(char* __restrict__ ws) {
    const uint4* hswz  = (const uint4*)(ws + WS_HSWZ);
    const uint4* wlswz = (const uint4*)(ws + WS_WLSWZ);
    const float* blin  = (const float*)(ws + WS_BLIN);
    float*       em    = (float*)(ws + WS_EM);

    const int m = blockIdx.x >> 6;
    const int grp = blockIdx.x & 63;
    const int wave = threadIdx.x >> 5;
    const int lane = threadIdx.x & 31;
    const int rowTile = grp * 16 + wave;     // 0..1023
    const int t = rowTile >> 3;
    const int mt = rowTile & 7;
    const int l16 = lane & 15;

    // stage this model's W_lin fragments (64KB) into LDS once
    async_copy_u4(wlswz + (size_t)(m * 64) * 64, 0, EMIS_W_U4);
    __syncthreads();

    v8f acc[4];
    #pragma unroll
    for (int st = 0; st < 4; ++st) {
        float bb = blin[m * S_SZ + st * 16 + l16];
        #pragma unroll
        for (int v = 0; v < 8; ++v) acc[st][v] = bb;
    }

    for (int kt = 0; kt < 16; ++kt) {
        v16bf af = load_frag(hswz, ((m * T_SZ + t) * 16 + kt) * 8 + mt, lane);
        #pragma unroll
        for (int st = 0; st < 4; ++st) {
            const int bi = (kt * 4 + st) * 64 + lane * 2;
            FragU bf;
            bf.u[0] = dynbuf[bi]; bf.u[1] = dynbuf[bi + 1];
            acc[st] = wmma_bf16(af, bf.v, acc[st]);
        }
    }

    // per-row log_softmax: 4 accs (same lane) + xor-reduce over the 16 lanes
    // of this half-wave (masks 1,2,4,8 keep halves independent).
    #pragma unroll
    for (int v = 0; v < 8; ++v) {
        float mx = fmaxf(fmaxf(acc[0][v], acc[1][v]), fmaxf(acc[2][v], acc[3][v]));
        #pragma unroll
        for (int off = 1; off < 16; off <<= 1)
            mx = fmaxf(mx, __shfl_xor(mx, off, 32));
        float sm = 0.f;
        #pragma unroll
        for (int st = 0; st < 4; ++st) sm += __expf(acc[st][v] - mx);
        #pragma unroll
        for (int off = 1; off < 16; off <<= 1)
            sm += __shfl_xor(sm, off, 32);
        float lse = mx + __logf(sm);

        int b = mt * 16 + v + 8 * (lane >> 4);
        size_t base = (((size_t)m * T_SZ + t) * B_SZ + b) * S_SZ + l16;
        #pragma unroll
        for (int st = 0; st < 4; ++st)
            em[base + st * 16] = acc[st][v] - lse;
    }
}

// ---------------------------------------------------------------------------
// Kernel 4: HMM forward recursion + final logsumexp. grid = 3*4 WGs,
// each WG: 32 batch rows; thread = (bloc, igroup of 4 states).
// ---------------------------------------------------------------------------
__global__ __launch_bounds__(512) void hmm_kernel(char* __restrict__ ws,
                                                  float* __restrict__ out) {
    const float* em     = (const float*)(ws + WS_EM);
    const float* logT   = (const float*)(ws + WS_LOGT);
    const float* logpri = (const float*)(ws + WS_LOGPRI);

    const int m = blockIdx.x >> 2;
    const int bg = blockIdx.x & 3;
    const int tid = threadIdx.x;
    const int bloc = tid >> 4;
    const int ig = tid & 15;
    const int b = bg * 32 + bloc;

    __shared__ float lt[64 * 65];        // row-padded to avoid bank conflicts
    __shared__ float alpha[32 * 64];

    for (int idx = tid; idx < 4096; idx += 512) {
        int i = idx >> 6, j = idx & 63;
        lt[i * 65 + j] = logT[m * 4096 + idx];
    }
    #pragma unroll
    for (int q = 0; q < 4; ++q) {
        int i = ig * 4 + q;
        alpha[bloc * 64 + i] =
            em[(((size_t)m * T_SZ + 0) * B_SZ + b) * S_SZ + i] + logpri[m * S_SZ + i];
    }
    __syncthreads();

    for (int t = 1; t < T_SZ; ++t) {
        float mx = -3.0e38f;
        for (int j = 0; j < 64; ++j) mx = fmaxf(mx, alpha[bloc * 64 + j]);
        float nv[4];
        #pragma unroll
        for (int q = 0; q < 4; ++q) {
            int i = ig * 4 + q;
            float s = 0.f;
            for (int j = 0; j < 64; ++j)
                s += __expf(lt[i * 65 + j] + alpha[bloc * 64 + j] - mx);
            nv[q] = em[(((size_t)m * T_SZ + t) * B_SZ + b) * S_SZ + i]
                    + mx + __logf(s);
        }
        __syncthreads();
        #pragma unroll
        for (int q = 0; q < 4; ++q) alpha[bloc * 64 + ig * 4 + q] = nv[q];
        __syncthreads();
    }

    if (ig == 0) {   // final logsumexp over states -> out[b][m]
        float mx = -3.0e38f;
        for (int j = 0; j < 64; ++j) mx = fmaxf(mx, alpha[bloc * 64 + j]);
        float s = 0.f;
        for (int j = 0; j < 64; ++j) s += __expf(alpha[bloc * 64 + j] - mx);
        out[b * M_SZ + m] = mx + __logf(s);
    }
}

// ---------------------------------------------------------------------------
extern "C" void kernel_launch(void* const* d_in, const int* in_sizes, int n_in,
                              void* d_out, int out_size, void* d_ws, size_t ws_size,
                              hipStream_t stream) {
    const float* x     = (const float*)d_in[0];
    const float* W_ih  = (const float*)d_in[1];
    const float* W_hh  = (const float*)d_in[2];
    const float* b_ih  = (const float*)d_in[3];
    const float* b_hh  = (const float*)d_in[4];
    const float* W_lin = (const float*)d_in[5];
    const float* b_lin = (const float*)d_in[6];
    const float* A     = (const float*)d_in[7];
    const float* prior = (const float*)d_in[8];
    char*  ws  = (char*)d_ws;
    float* out = (float*)d_out;

    (void)in_sizes; (void)n_in; (void)out_size; (void)ws_size;

    // allow >64KB dynamic LDS for the staged-weight kernels (not stream-ordered,
    // safe under graph capture)
    (void)hipFuncSetAttribute((const void*)lstm_kernel,
                              hipFuncAttributeMaxDynamicSharedMemorySize,
                              LSTM_LDS_BYTES);
    (void)hipFuncSetAttribute((const void*)emis_kernel,
                              hipFuncAttributeMaxDynamicSharedMemorySize,
                              EMIS_LDS_BYTES);

    int prepBlocks = (PREP_TOTAL + 255) / 256;
    prep_kernel<<<prepBlocks, 256, 0, stream>>>(x, W_ih, W_hh, b_ih, b_hh,
                                                W_lin, b_lin, A, prior, ws);
    lstm_kernel<<<M_SZ * NSPLIT, 512, LSTM_LDS_BYTES, stream>>>(ws);
    emis_kernel<<<M_SZ * 64, 512, EMIS_LDS_BYTES, stream>>>(ws);
    hmm_kernel<<<M_SZ * 4, 512, 0, stream>>>(ws, out);
}
